// Encoder_55362128445616
// MI455X (gfx1250) — compile-verified, matrix operand order checked by reading, not run on hardware
//
#include <hip/hip_runtime.h>
#include <hip/hip_bf16.h>
#include <cstdint>
#include <cstddef>

// ---------------------------------------------------------------------------
// Transformer encoder layer on gfx1250 (MI455X), bf16 WMMA + fp32 accumulate.
// GEMM staging uses the Tensor Data Mover (tensor_load_to_lds + TENSORcnt)
// when the builtin is available; falls back to vector global->LDS staging.
// ---------------------------------------------------------------------------

typedef __attribute__((ext_vector_type(16))) __bf16 bf16x16;
typedef __attribute__((ext_vector_type(8)))  float  f32x8;
typedef __attribute__((ext_vector_type(4)))  unsigned int u32x4_t;
typedef __attribute__((ext_vector_type(8)))  int i32x8_t;
typedef __attribute__((ext_vector_type(4)))  int i32x4_t;

#define BM 128
#define BN 128
#define BK 32
#define LDS_K 40          // 32 + 8 halfs padding (80B row stride, 16B aligned)
#define TPB 256           // 8 wave32 waves

#if defined(__has_builtin)
#  if __has_builtin(__builtin_amdgcn_tensor_load_to_lds)
#    define HAVE_TDM 1
#  endif
#endif
#ifndef HAVE_TDM
#  define HAVE_TDM 0
#endif

union FragU {
    bf16x16 v;
    uint4   q[2];
};

#if HAVE_TDM
__device__ __forceinline__ void wait_tensorcnt0()
{
#if __has_builtin(__builtin_amdgcn_s_wait_tensorcnt)
    __builtin_amdgcn_s_wait_tensorcnt(0);
#else
    asm volatile("s_wait_tensorcnt 0x0" ::: "memory");
#endif
}

// Issue one TDM transfer of a [128 rows x 32 elems] bf16 tile into LDS.
// LDS destination gets 16B of padding after every 64B row (row stride 80B),
// matching the sA/sB [..][LDS_K] layout used by the WMMA fragment loads.
//   D# group0: count=1, lds_addr, global_addr(57b), type=2
//   D# group1: data_size=2B, pad_enable, pad_interval=16 DW, pad_amount=4 DW,
//              tensor_dim0=K, tensor_dim1=huge, tile=32x128, dim0_stride=K
__device__ __forceinline__ void tdm_load_tile(unsigned int lds_addr,
                                              const __bf16* gptr, int K)
{
    const unsigned long long ga = (unsigned long long)(uintptr_t)gptr;
    u32x4_t g0;
    g0[0] = 1u;                                    // count=1 (valid), user mode
    g0[1] = lds_addr;                              // LDS byte address
    g0[2] = (unsigned int)ga;                      // global_addr[31:0]
    g0[3] = (unsigned int)((ga >> 32) & 0x01FFFFFFu) | (2u << 30); // [56:32]|type=2

    const unsigned int td0 = (unsigned int)K;      // tensor dim0 (elements)
    const unsigned int td1 = 0x00100000u;          // big: tiles never go OOB
    const unsigned long long st0 = (unsigned long long)K; // dim0 stride (elems)

    i32x8_t g1;
    g1[0] = (int)((1u << 16)        // data_size = 1 -> 2 bytes
                | (1u << 20)        // pad_enable
                | (3u << 22)        // pad_interval: 2^(3+1) = 16 DWORDs = 64 B
                | (3u << 25));      // pad_amount: 4 DWORDs = 16 B
    g1[1] = (int)((td0 & 0xFFFFu) << 16);                       // dim0 lo16
    g1[2] = (int)(((td0 >> 16) & 0xFFFFu) | ((td1 & 0xFFFFu) << 16));
    g1[3] = (int)(((td1 >> 16) & 0xFFFFu) | (32u << 16));       // tile_dim0=32
    g1[4] = 128;                                                // tile_dim1=128
    g1[5] = (int)(st0 & 0xFFFFFFFFu);                           // stride0 lo32
    g1[6] = (int)((st0 >> 32) & 0xFFFFu);                       // stride0 hi16
    g1[7] = 0;

    const i32x4_t z4 = {0, 0, 0, 0};               // groups 2/3: 2D tensor
    const i32x8_t z8 = {0, 0, 0, 0, 0, 0, 0, 0};
    // 6-arg form (amdgpu-toolchain / clang-23 HIP headers)
    __builtin_amdgcn_tensor_load_to_lds(g0, g1, z4, z4, z8, 0);
}
#endif // HAVE_TDM

// ---------------------------------------------------------------------------
// Generic bf16 WMMA GEMM:  C = act( scale * (A @ B^T) + bias )
//   A: [M,K] bf16 row-major (stride sA/batch), B: [N,K] bf16 row-major
//   bias fp32 indexed by col (BIAS_ROW=false) or row (BIAS_ROW=true)
// M,N multiples of 128; K multiple of 32.
// ---------------------------------------------------------------------------
template <typename OutT, bool RELU, bool BIAS_ROW>
__global__ __launch_bounds__(TPB)
void gemm_bf16_wmma(const __bf16* __restrict__ A, const __bf16* __restrict__ B,
                    OutT* __restrict__ C, const float* __restrict__ bias,
                    int M, int N, int K,
                    long long strideA, long long strideB, long long strideC,
                    float scale)
{
    const int tid  = threadIdx.x;
    const int lane = tid & 31;
    const int wid  = tid >> 5;
    const int wm   = wid & 3;          // 4 waves along M  -> 32 rows each
    const int wn   = wid >> 2;         // 2 waves along N  -> 64 cols each
    const int r    = lane & 15;
    const int hk   = (lane >> 4) << 3; // lanes 16..31 own K+8 groups

    const int blockRow = blockIdx.y * BM;
    const int blockCol = blockIdx.x * BN;

    const __bf16* Ag = A + (size_t)blockIdx.z * strideA;
    const __bf16* Bg = B + (size_t)blockIdx.z * strideB;
    OutT*         Cg = C + (size_t)blockIdx.z * strideC;

    f32x8 acc[2][4];
#pragma unroll
    for (int i = 0; i < 2; ++i)
#pragma unroll
        for (int j = 0; j < 4; ++j)
            acc[i][j] = (f32x8)0.0f;

#if HAVE_TDM
    // ---------------- TDM-staged, double-buffered main loop ----------------
    __shared__ __bf16 sA[2][BM][LDS_K];
    __shared__ __bf16 sB[2][BN][LDS_K];

    const unsigned int ldsA0 = (unsigned int)(uintptr_t)&sA[0][0][0];
    const unsigned int ldsA1 = (unsigned int)(uintptr_t)&sA[1][0][0];
    const unsigned int ldsB0 = (unsigned int)(uintptr_t)&sB[0][0][0];
    const unsigned int ldsB1 = (unsigned int)(uintptr_t)&sB[1][0][0];

    const int T = K / BK;
    if (wid == 0) {
        tdm_load_tile(ldsA0, Ag + (size_t)blockRow * K, K);
        tdm_load_tile(ldsB0, Bg + (size_t)blockCol * K, K);
    }

    for (int t = 0; t < T; ++t) {
        const int buf = t & 1;
        if (wid == 0) wait_tensorcnt0();     // tile t fully in LDS
        __syncthreads();                     // publish tile t to all waves

        // overlap: DMA tile t+1 into the other buffer while computing tile t
        if (wid == 0 && (t + 1) < T) {
            const size_t k1 = (size_t)(t + 1) * BK;
            tdm_load_tile(buf ? ldsA0 : ldsA1, Ag + (size_t)blockRow * K + k1, K);
            tdm_load_tile(buf ? ldsB0 : ldsB1, Bg + (size_t)blockCol * K + k1, K);
        }

        FragU af[2], bfr[4];
#pragma unroll
        for (int i = 0; i < 2; ++i) {
            const __bf16* p = &sA[buf][wm * 32 + i * 16 + r][0];
            af[i].q[0] = *(const uint4*)(p + hk);
            af[i].q[1] = *(const uint4*)(p + 16 + hk);
        }
#pragma unroll
        for (int j = 0; j < 4; ++j) {
            const __bf16* p = &sB[buf][wn * 64 + j * 16 + r][0];
            bfr[j].q[0] = *(const uint4*)(p + hk);
            bfr[j].q[1] = *(const uint4*)(p + 16 + hk);
        }
#pragma unroll
        for (int i = 0; i < 2; ++i)
#pragma unroll
            for (int j = 0; j < 4; ++j)
                acc[i][j] = __builtin_amdgcn_wmma_f32_16x16x32_bf16(
                    false, af[i].v, false, bfr[j].v,
                    (short)0, acc[i][j], false, false);

        __syncthreads();                     // all done reading buffer `buf`
    }
#else
    // ---------------- fallback: vector global->LDS staging -----------------
    __shared__ __bf16 sA[1][BM][LDS_K];
    __shared__ __bf16 sB[1][BN][LDS_K];

    const int srow = tid >> 2;               // 0..63
    const int skc  = (tid & 3) << 3;         // 0,8,16,24 halfs

    for (int k0 = 0; k0 < K; k0 += BK) {
        *(uint4*)&sA[0][srow][skc] =
            *(const uint4*)(Ag + (size_t)(blockRow + srow) * K + k0 + skc);
        *(uint4*)&sA[0][srow + 64][skc] =
            *(const uint4*)(Ag + (size_t)(blockRow + srow + 64) * K + k0 + skc);
        *(uint4*)&sB[0][srow][skc] =
            *(const uint4*)(Bg + (size_t)(blockCol + srow) * K + k0 + skc);
        *(uint4*)&sB[0][srow + 64][skc] =
            *(const uint4*)(Bg + (size_t)(blockCol + srow + 64) * K + k0 + skc);

        if (k0 + BK < K) {
            __builtin_prefetch(Ag + (size_t)(blockRow + srow) * K + k0 + BK + skc, 0, 1);
            __builtin_prefetch(Bg + (size_t)(blockCol + srow) * K + k0 + BK + skc, 0, 1);
        }
        __syncthreads();

        FragU af[2], bfr[4];
#pragma unroll
        for (int i = 0; i < 2; ++i) {
            const __bf16* p = &sA[0][wm * 32 + i * 16 + r][0];
            af[i].q[0] = *(const uint4*)(p + hk);
            af[i].q[1] = *(const uint4*)(p + 16 + hk);
        }
#pragma unroll
        for (int j = 0; j < 4; ++j) {
            const __bf16* p = &sB[0][wn * 64 + j * 16 + r][0];
            bfr[j].q[0] = *(const uint4*)(p + hk);
            bfr[j].q[1] = *(const uint4*)(p + 16 + hk);
        }
#pragma unroll
        for (int i = 0; i < 2; ++i)
#pragma unroll
            for (int j = 0; j < 4; ++j)
                acc[i][j] = __builtin_amdgcn_wmma_f32_16x16x32_bf16(
                    false, af[i].v, false, bfr[j].v,
                    (short)0, acc[i][j], false, false);

        __syncthreads();
    }
#endif

    // ---- epilogue: C layout  m = vgpr + 8*(lane>=16), n = lane&15
    const int mb = (lane >> 4) << 3;
    const int n0 = lane & 15;
#pragma unroll
    for (int i = 0; i < 2; ++i)
#pragma unroll
        for (int j = 0; j < 4; ++j)
#pragma unroll
            for (int v = 0; v < 8; ++v) {
                const int row = blockRow + wm * 32 + i * 16 + mb + v;
                const int col = blockCol + wn * 64 + j * 16 + n0;
                float val = acc[i][j][v] * scale;
                if (bias) val += BIAS_ROW ? bias[row] : bias[col];
                if (RELU) val = fmaxf(val, 0.0f);
                Cg[(size_t)row * N + col] = (OutT)val;
            }
}

// ---------------------------------------------------------------------------
// fp32 -> bf16 conversion
// ---------------------------------------------------------------------------
__global__ void cvt_f32_bf16(const float* __restrict__ in, __bf16* __restrict__ out, int n)
{
    int i = blockIdx.x * blockDim.x + threadIdx.x;
    if (i < n) out[i] = (__bf16)in[i];
}

// ---------------------------------------------------------------------------
// Row softmax (1024 cols), fp32 in -> bf16 out. One block (256 thr) per row.
// ---------------------------------------------------------------------------
__global__ __launch_bounds__(256)
void softmax_rows(const float* __restrict__ S, __bf16* __restrict__ P)
{
    const float* row = S + (size_t)blockIdx.x * 1024;
    __bf16*     prow = P + (size_t)blockIdx.x * 1024;
    const int t = threadIdx.x;

    float v0 = row[t], v1 = row[t + 256], v2 = row[t + 512], v3 = row[t + 768];

    __shared__ float sm[8];
    __shared__ float red[2];

    float mx = fmaxf(fmaxf(v0, v1), fmaxf(v2, v3));
#pragma unroll
    for (int o = 16; o; o >>= 1) mx = fmaxf(mx, __shfl_xor(mx, o, 32));
    if ((t & 31) == 0) sm[t >> 5] = mx;
    __syncthreads();
    if (t == 0) {
        float m = sm[0];
        for (int i = 1; i < 8; ++i) m = fmaxf(m, sm[i]);
        red[0] = m;
    }
    __syncthreads();
    mx = red[0];

    float e0 = __expf(v0 - mx), e1 = __expf(v1 - mx);
    float e2 = __expf(v2 - mx), e3 = __expf(v3 - mx);
    float s = e0 + e1 + e2 + e3;
#pragma unroll
    for (int o = 16; o; o >>= 1) s += __shfl_xor(s, o, 32);
    if ((t & 31) == 0) sm[t >> 5] = s;
    __syncthreads();
    if (t == 0) {
        float z = 0.0f;
        for (int i = 0; i < 8; ++i) z += sm[i];
        red[1] = 1.0f / z;
    }
    __syncthreads();
    const float rinv = red[1];

    prow[t]       = (__bf16)(e0 * rinv);
    prow[t + 256] = (__bf16)(e1 * rinv);
    prow[t + 512] = (__bf16)(e2 * rinv);
    prow[t + 768] = (__bf16)(e3 * rinv);
}

// ---------------------------------------------------------------------------
// out = LayerNorm(a + b) * g + be   (row length 512; one block per row)
// Optionally also writes a bf16 copy (feeds the next GEMM).
// ---------------------------------------------------------------------------
__global__ __launch_bounds__(256)
void add_ln(const float* __restrict__ a, const float* __restrict__ b,
            const float* __restrict__ g, const float* __restrict__ be,
            float* __restrict__ of, __bf16* __restrict__ ob)
{
    const size_t base = (size_t)blockIdx.x * 512;
    const int t = threadIdx.x;

    float x0 = a[base + t]       + b[base + t];
    float x1 = a[base + t + 256] + b[base + t + 256];

    float s  = x0 + x1;
    float sq = x0 * x0 + x1 * x1;
#pragma unroll
    for (int o = 16; o; o >>= 1) {
        s  += __shfl_xor(s, o, 32);
        sq += __shfl_xor(sq, o, 32);
    }
    __shared__ float ss[8], qq[8];
    __shared__ float red[2];
    if ((t & 31) == 0) { ss[t >> 5] = s; qq[t >> 5] = sq; }
    __syncthreads();
    if (t == 0) {
        float S = 0.0f, Q = 0.0f;
        for (int i = 0; i < 8; ++i) { S += ss[i]; Q += qq[i]; }
        red[0] = S; red[1] = Q;
    }
    __syncthreads();
    const float m    = red[0] * (1.0f / 512.0f);
    const float var  = red[1] * (1.0f / 512.0f) - m * m;
    const float rinv = rsqrtf(var + 1e-5f);

    float y0 = (x0 - m) * rinv * g[t]       + be[t];
    float y1 = (x1 - m) * rinv * g[t + 256] + be[t + 256];
    of[base + t]       = y0;
    of[base + t + 256] = y1;
    if (ob) {
        ob[base + t]       = (__bf16)y0;
        ob[base + t + 256] = (__bf16)y1;
    }
}

// ---------------------------------------------------------------------------
// Host-side orchestration
// ---------------------------------------------------------------------------
static inline char* carve(char*& w, size_t bytes)
{
    char* p = w;
    w += (bytes + 255) & ~(size_t)255;
    return p;
}

extern "C" void kernel_launch(void* const* d_in, const int* in_sizes, int n_in,
                              void* d_out, int out_size, void* d_ws, size_t ws_size,
                              hipStream_t stream)
{
    (void)in_sizes; (void)n_in; (void)out_size; (void)ws_size;

    const int Bb = 8, S = 1024, D = 512, E = 4096;
    const int M  = Bb * S;                 // 8192 flattened rows
    const float inv_sqrt_d = 0.044194173824159216f; // 1/sqrt(512)

    const float* x  = (const float*)d_in[0];
    const float* Wq = (const float*)d_in[1];
    const float* bq = (const float*)d_in[2];
    const float* Wk = (const float*)d_in[3];
    const float* bk = (const float*)d_in[4];
    const float* Wv = (const float*)d_in[5];
    const float* bv = (const float*)d_in[6];
    const float* Wo = (const float*)d_in[7];
    const float* bo = (const float*)d_in[8];
    const float* g0 = (const float*)d_in[9];
    const float* be0= (const float*)d_in[10];
    const float* W1 = (const float*)d_in[11];
    const float* b1 = (const float*)d_in[12];
    const float* W2 = (const float*)d_in[13];
    const float* b2 = (const float*)d_in[14];
    const float* g1 = (const float*)d_in[15];
    const float* be1= (const float*)d_in[16];
    float* out = (float*)d_out;

    // workspace carve-up
    char* w = (char*)d_ws;
    __bf16* xb   = (__bf16*)carve(w, (size_t)M * D * 2);
    __bf16* Wqb  = (__bf16*)carve(w, (size_t)E * D * 2);
    __bf16* Wkb  = (__bf16*)carve(w, (size_t)E * D * 2);
    __bf16* Wvb  = (__bf16*)carve(w, (size_t)E * D * 2);
    __bf16* Wob  = (__bf16*)carve(w, (size_t)D * E * 2);
    __bf16* W1b  = (__bf16*)carve(w, (size_t)D * D * 2);
    __bf16* W2b  = (__bf16*)carve(w, (size_t)D * D * 2);
    __bf16* Qb   = (__bf16*)carve(w, (size_t)M * E * 2);
    __bf16* Kb   = (__bf16*)carve(w, (size_t)M * E * 2);
    __bf16* Vtb  = (__bf16*)carve(w, (size_t)Bb * E * S * 2);   // V^T per batch
    float*  scr  = (float*) carve(w, (size_t)Bb * S * S * 4);
    __bf16* attn = (__bf16*)carve(w, (size_t)Bb * S * S * 2);
    __bf16* sdp  = (__bf16*)carve(w, (size_t)M * E * 2);
    float*  mha  = (float*) carve(w, (size_t)M * D * 4);
    float*  h    = (float*) carve(w, (size_t)M * D * 4);
    __bf16* hb   = (__bf16*)carve(w, (size_t)M * D * 2);
    __bf16* ff1  = (__bf16*)carve(w, (size_t)M * D * 2);
    float*  ff2  = (float*) carve(w, (size_t)M * D * 4);

    // ---- fp32 -> bf16 conversions
    auto cvt = [&](const float* src, __bf16* dst, int n) {
        cvt_f32_bf16<<<(n + 255) / 256, 256, 0, stream>>>(src, dst, n);
    };
    cvt(x,  xb,  M * D);
    cvt(Wq, Wqb, E * D);
    cvt(Wk, Wkb, E * D);
    cvt(Wv, Wvb, E * D);
    cvt(Wo, Wob, D * E);
    cvt(W1, W1b, D * D);
    cvt(W2, W2b, D * D);

    // ---- Q = x @ Wq^T + bq   [8192, 4096] bf16
    gemm_bf16_wmma<__bf16, false, false>
        <<<dim3(E / BN, M / BM, 1), TPB, 0, stream>>>(
            xb, Wqb, Qb, bq, M, E, D, 0, 0, 0, 1.0f);
    // ---- K = x @ Wk^T + bk
    gemm_bf16_wmma<__bf16, false, false>
        <<<dim3(E / BN, M / BM, 1), TPB, 0, stream>>>(
            xb, Wkb, Kb, bk, M, E, D, 0, 0, 0, 1.0f);
    // ---- V^T[b] = Wv @ x[b]^T + bv(row)   [8, 4096, 1024] bf16
    gemm_bf16_wmma<__bf16, false, true>
        <<<dim3(S / BN, E / BM, Bb), TPB, 0, stream>>>(
            Wvb, xb, Vtb, bv, E, S, D,
            0, (long long)S * D, (long long)E * S, 1.0f);

    // ---- scores[b] = (Q[b] @ K[b]^T) / sqrt(512)   [8, 1024, 1024] fp32
    gemm_bf16_wmma<float, false, false>
        <<<dim3(S / BN, S / BM, Bb), TPB, 0, stream>>>(
            Qb, Kb, scr, nullptr, S, S, E,
            (long long)S * E, (long long)S * E, (long long)S * S, inv_sqrt_d);

    // ---- attn = softmax(scores) -> bf16
    softmax_rows<<<Bb * S, 256, 0, stream>>>(scr, attn);

    // ---- sdpa[b] = attn[b] @ (V^T[b])^T   [8, 1024, 4096] bf16
    gemm_bf16_wmma<__bf16, false, false>
        <<<dim3(E / BN, S / BM, Bb), TPB, 0, stream>>>(
            attn, Vtb, sdp, nullptr, S, E, S,
            (long long)S * S, (long long)E * S, (long long)S * E, 1.0f);

    // ---- mha = sdpa @ Wo^T + bo   [8192, 512] fp32
    gemm_bf16_wmma<float, false, false>
        <<<dim3(D / BN, M / BM, 1), TPB, 0, stream>>>(
            sdp, Wob, mha, bo, M, D, E, 0, 0, 0, 1.0f);

    // ---- h = LN(mha + x) * g0 + be0  (fp32 + bf16 copy)
    add_ln<<<M, 256, 0, stream>>>(mha, x, g0, be0, h, hb);

    // ---- ff1 = relu(h @ W1^T + b1)  bf16
    gemm_bf16_wmma<__bf16, true, false>
        <<<dim3(D / BN, M / BM, 1), TPB, 0, stream>>>(
            hb, W1b, ff1, b1, M, D, D, 0, 0, 0, 1.0f);

    // ---- ff2 = ff1 @ W2^T + b2  fp32
    gemm_bf16_wmma<float, false, false>
        <<<dim3(D / BN, M / BM, 1), TPB, 0, stream>>>(
            ff1, W2b, ff2, b2, M, D, D, 0, 0, 0, 1.0f);

    // ---- out = LN(ff2 + h) * g1 + be1  fp32
    add_ln<<<M, 256, 0, stream>>>(ff2, h, g1, be1, out, nullptr);
}
